// ToME2D_69045894250787
// MI455X (gfx1250) — compile-verified, matrix operand order checked by reading, not run on hardware
//
#include <hip/hip_runtime.h>
#include <hip/hip_bf16.h>

typedef __attribute__((ext_vector_type(16))) __bf16 v16bf;
typedef __attribute__((ext_vector_type(8)))  __bf16 v8bf;
typedef __attribute__((ext_vector_type(4)))  __bf16 v4bf;
typedef __attribute__((ext_vector_type(8)))  float  v8f;

#define B_   64
#define N_   1024
#define C_   1152
#define NSRC 768
#define NDST 256
#define R_   512
#define NUNM 256

// ---- deterministic ToMe 2x2 partition (h=w=32, sx=sy=2), row-major stable order ----
// dst tokens: even row, even col. src tokens: everything else, row-major.
__device__ __forceinline__ int a_token(int s) {
    // 2-row group holds 48 src: 16 at odd x in even row, 32 in odd row
    int g = s / 48, rem = s - g * 48;
    int y, x;
    if (rem < 16) { y = 2 * g;     x = 2 * rem + 1; }
    else          { y = 2 * g + 1; x = rem - 16;    }
    return y * 32 + x;
}
__device__ __forceinline__ int b_token(int d) {
    return (2 * (d >> 4)) * 32 + 2 * (d & 15);
}

// ---- 1 / ||x_row||  (one wave per token row) ----
__global__ void __launch_bounds__(256) k_norm(const float* __restrict__ x,
                                              float* __restrict__ inv_norm) {
    int wave = threadIdx.x >> 5, lane = threadIdx.x & 31;
    int row = blockIdx.x * 8 + wave;            // 0 .. B_*N_-1
    const float* p = x + (size_t)row * C_;
    float s = 0.f;
    for (int c = lane * 4; c < C_; c += 128) {
        float4 v = *(const float4*)(p + c);
        s += v.x * v.x + v.y * v.y + v.z * v.z + v.w * v.w;
    }
    #pragma unroll
    for (int off = 16; off > 0; off >>= 1) s += __shfl_xor(s, off, 32);
    if (lane == 0) inv_norm[row] = 1.0f / sqrtf(s);
}

// ---- dst metric panel, normalized, bf16 ----
__global__ void __launch_bounds__(288) k_metric_b(const float* __restrict__ x,
                                                  const float* __restrict__ inv_norm,
                                                  __bf16* __restrict__ mb) {
    int d = blockIdx.x, b = blockIdx.y, t = threadIdx.x;   // 288 threads * 4 floats
    int tok = b_token(d);
    const float* p = x + ((size_t)b * N_ + tok) * C_;
    float inv = inv_norm[b * N_ + tok];
    float4 v = *(const float4*)(p + t * 4);
    v4bf h;
    h[0] = (__bf16)(v.x * inv); h[1] = (__bf16)(v.y * inv);
    h[2] = (__bf16)(v.z * inv); h[3] = (__bf16)(v.w * inv);
    *(v4bf*)(mb + ((size_t)b * NDST + d) * C_ + t * 4) = h;
}

// ---- scores GEMM (bf16 WMMA) fused with row max/argmax ----
// wave -> 16 src rows x 256 dst cols, K=1152 in steps of 32, 16 live accumulators.
// B fragments double-buffered; A f32 row loads pipelined one k-step ahead.
__global__ void __launch_bounds__(256) k_gemm(const float* __restrict__ x,
                                              const float* __restrict__ inv_norm,
                                              const __bf16* __restrict__ mb,
                                              float* __restrict__ node_max,
                                              int* __restrict__ node_idx) {
    int b = blockIdx.y;
    int wave = threadIdx.x >> 5, lane = threadIdx.x & 31;
    int l16 = lane & 15, hi = lane >> 4;
    int m0 = (blockIdx.x * 8 + wave) * 16;

    int tokA = a_token(m0 + l16);
    const float* arow = x + ((size_t)b * N_ + tokA) * C_;
    float inv = inv_norm[b * N_ + tokA];
    const __bf16* bbase = mb + ((size_t)b * NDST + l16) * C_;

    v8f acc[16];
    #pragma unroll
    for (int nt = 0; nt < 16; ++nt) acc[nt] = (v8f)(0.f);

    // prime A-row loads for k0 = 0
    int ka = hi * 8;            // k0 + hi*8
    int kb = 16 + hi * 8;       // k0 + 16 + hi*8
    float4 f0 = *(const float4*)(arow + ka);
    float4 f1 = *(const float4*)(arow + ka + 4);
    float4 f2 = *(const float4*)(arow + kb);
    float4 f3 = *(const float4*)(arow + kb + 4);

    for (int k0 = 0; k0 < C_; k0 += 32) {
        // build A fragment from the in-flight f32 data
        v16bf A;
        A[0]  = (__bf16)(f0.x * inv); A[1]  = (__bf16)(f0.y * inv);
        A[2]  = (__bf16)(f0.z * inv); A[3]  = (__bf16)(f0.w * inv);
        A[4]  = (__bf16)(f1.x * inv); A[5]  = (__bf16)(f1.y * inv);
        A[6]  = (__bf16)(f1.z * inv); A[7]  = (__bf16)(f1.w * inv);
        A[8]  = (__bf16)(f2.x * inv); A[9]  = (__bf16)(f2.y * inv);
        A[10] = (__bf16)(f2.z * inv); A[11] = (__bf16)(f2.w * inv);
        A[12] = (__bf16)(f3.x * inv); A[13] = (__bf16)(f3.y * inv);
        A[14] = (__bf16)(f3.z * inv); A[15] = (__bf16)(f3.w * inv);

        // issue next k-step's A loads so they fly under the 16 WMMAs
        if (k0 + 32 < C_) {
            int ka2 = k0 + 32 + hi * 8;
            int kb2 = k0 + 48 + hi * 8;
            f0 = *(const float4*)(arow + ka2);
            f1 = *(const float4*)(arow + ka2 + 4);
            f2 = *(const float4*)(arow + kb2);
            f3 = *(const float4*)(arow + kb2 + 4);
        }

        // B fragments, double-buffered: prefetch nt+1 before wmma(nt)
        v8bf blo0 = *(const v8bf*)(bbase + k0 + (hi * 8));
        v8bf bhi0 = *(const v8bf*)(bbase + k0 + 16 + (hi * 8));
        #pragma unroll
        for (int nt = 0; nt < 16; ++nt) {
            v8bf blo1 = blo0, bhi1 = bhi0;
            if (nt < 15) {
                const __bf16* bp = bbase + (size_t)(nt + 1) * 16 * C_;
                blo1 = *(const v8bf*)(bp + k0 + (hi * 8));
                bhi1 = *(const v8bf*)(bp + k0 + 16 + (hi * 8));
            }
            v16bf Bf;
            #pragma unroll
            for (int i = 0; i < 8; ++i) { Bf[i] = blo0[i]; Bf[i + 8] = bhi0[i]; }
            acc[nt] = __builtin_amdgcn_wmma_f32_16x16x32_bf16(
                false, A, false, Bf, (short)0, acc[nt], false, false);
            blo0 = blo1; bhi0 = bhi1;
        }
    }

    // running max/argmax per (row, lane-col), first-max tie break
    float best[8]; int bidx[8];
    #pragma unroll
    for (int v = 0; v < 8; ++v) { best[v] = -3.4e38f; bidx[v] = 0; }
    #pragma unroll
    for (int nt = 0; nt < 16; ++nt) {
        #pragma unroll
        for (int v = 0; v < 8; ++v) {
            float val = acc[nt][v];
            if (val > best[v]) { best[v] = val; bidx[v] = nt * 16 + l16; }
        }
    }
    // reduce across the 16 lanes of each half (rows m0+v and m0+8+v)
    #pragma unroll
    for (int v = 0; v < 8; ++v) {
        #pragma unroll
        for (int off = 1; off < 16; off <<= 1) {
            float ov = __shfl_xor(best[v], off, 32);
            int   oi = __shfl_xor(bidx[v], off, 32);
            if (ov > best[v] || (ov == best[v] && oi < bidx[v])) { best[v] = ov; bidx[v] = oi; }
        }
        if (l16 == 0) {
            int row = m0 + v + hi * 8;
            node_max[(size_t)b * NSRC + row] = best[v];
            node_idx[(size_t)b * NSRC + row] = bidx[v];
        }
    }
}

// ---- stable rank of -node_max (reproduces stable argsort), deterministic lists ----
__global__ void __launch_bounds__(NSRC) k_rank(const float* __restrict__ node_max,
                                               const int* __restrict__ node_idx,
                                               int* __restrict__ merged_src,
                                               int* __restrict__ merged_dst,
                                               int* __restrict__ unm_src) {
    __shared__ float vals[NSRC];
    int b = blockIdx.x, i = threadIdx.x;
    float vi = node_max[(size_t)b * NSRC + i];
    vals[i] = vi;
    __syncthreads();
    int rank = 0;
    for (int j = 0; j < NSRC; ++j) {
        float vj = vals[j];
        rank += (vj > vi) || (vj == vi && j < i);
    }
    if (rank < R_) {
        merged_src[b * R_ + rank] = i;
        merged_dst[b * R_ + rank] = node_idx[(size_t)b * NSRC + i];
    } else {
        unm_src[b * NUNM + (rank - R_)] = i;
    }
}

// ---- copy unmerged rows ----
__global__ void __launch_bounds__(288) k_unm(const float* __restrict__ x,
                                             const int* __restrict__ unm_src,
                                             float* __restrict__ out) {
    int u = blockIdx.x, b = blockIdx.y, t = threadIdx.x;
    int tok = a_token(unm_src[b * NUNM + u]);
    float4 v = *(const float4*)(x + ((size_t)b * N_ + tok) * C_ + t * 4);
    *(float4*)(out + ((size_t)b * (NUNM + NDST) + u) * C_ + t * 4) = v;
}

// ---- dst rows: deterministic rank-ordered merge average ----
__global__ void __launch_bounds__(288) k_dst(const float* __restrict__ x,
                                             const int* __restrict__ merged_src,
                                             const int* __restrict__ merged_dst,
                                             float* __restrict__ out) {
    int d = blockIdx.x, b = blockIdx.y, t = threadIdx.x;
    float4 acc = *(const float4*)(x + ((size_t)b * N_ + b_token(d)) * C_ + t * 4);
    int cnt = 0;
    for (int s = 0; s < R_; ++s) {
        if (merged_dst[b * R_ + s] == d) {
            int tok = a_token(merged_src[b * R_ + s]);
            float4 v = *(const float4*)(x + ((size_t)b * N_ + tok) * C_ + t * 4);
            acc.x += v.x; acc.y += v.y; acc.z += v.z; acc.w += v.w;
            ++cnt;
        }
    }
    float sc = 1.0f / (float)(cnt + 1);
    acc.x *= sc; acc.y *= sc; acc.z *= sc; acc.w *= sc;
    *(float4*)(out + ((size_t)b * (NUNM + NDST) + NUNM + d) * C_ + t * 4) = acc;
}

extern "C" void kernel_launch(void* const* d_in, const int* in_sizes, int n_in,
                              void* d_out, int out_size, void* d_ws, size_t ws_size,
                              hipStream_t stream) {
    const float* x = (const float*)d_in[0];   // (64,1024,1152) f32; scalar args hardcoded
    float* out = (float*)d_out;               // (64,512,1152) f32

    char* ws = (char*)d_ws;
    float*  inv_norm   = (float*)ws;  ws += (size_t)B_ * N_ * sizeof(float);
    __bf16* mb         = (__bf16*)ws; ws += (size_t)B_ * NDST * C_ * sizeof(__bf16);
    float*  node_max   = (float*)ws;  ws += (size_t)B_ * NSRC * sizeof(float);
    int*    node_idx   = (int*)ws;    ws += (size_t)B_ * NSRC * sizeof(int);
    int*    merged_src = (int*)ws;    ws += (size_t)B_ * R_ * sizeof(int);
    int*    merged_dst = (int*)ws;    ws += (size_t)B_ * R_ * sizeof(int);
    int*    unm_src    = (int*)ws;    ws += (size_t)B_ * NUNM * sizeof(int);

    k_norm    <<<dim3((B_ * N_) / 8), 256, 0, stream>>>(x, inv_norm);
    k_metric_b<<<dim3(NDST, B_),      288, 0, stream>>>(x, inv_norm, mb);
    k_gemm    <<<dim3(6, B_),         256, 0, stream>>>(x, inv_norm, mb, node_max, node_idx);
    k_rank    <<<dim3(B_),           NSRC, 0, stream>>>(node_max, node_idx,
                                                        merged_src, merged_dst, unm_src);
    k_unm     <<<dim3(NUNM, B_),      288, 0, stream>>>(x, unm_src, out);
    k_dst     <<<dim3(NDST, B_),      288, 0, stream>>>(x, merged_src, merged_dst, out);
}
